// SpaAttenLSTM_V2_71691594105147
// MI455X (gfx1250) — compile-verified, matrix operand order checked by reading, not run on hardware
//
#include <hip/hip_runtime.h>
#include <hip/hip_bf16.h>
#include <cmath>

// Problem sizes (compile-time)
#define B_ 8
#define T_ 256
#define S_ 196
#define F_ 512
#define H_ 128
#define G_ 512   // 4*H

typedef __attribute__((ext_vector_type(16))) _Float16 v16h;
typedef __attribute__((ext_vector_type(8)))  _Float16 v8h;
typedef __attribute__((ext_vector_type(8)))  float    v8f;
typedef __attribute__((ext_vector_type(4)))  float    v4f;

__device__ __forceinline__ float sigmoidf_(float x) { return 1.0f / (1.0f + __expf(-x)); }

__device__ __forceinline__ float wave_sum(float v) {
#pragma unroll
  for (int off = 16; off >= 1; off >>= 1) v += __shfl_xor(v, off, 32);
  return v;
}

// ---------------------------------------------------------------------------
// Kernel 1: fused spatial attention pooling with ONLINE softmax (single pass
// over x: 822 MB read once -> ~35 us HBM floor at 23.3 TB/s). One workgroup
// per (b,t); 8 waves each own a strided subset of the 196 spatial slots with
// private (m, l, acc[512]) running state; merged once at the end through LDS.
// No barriers in the hot loop; x streamed with NT hints + prefetch.
// ---------------------------------------------------------------------------
__global__ __launch_bounds__(256) void attn_pool_kernel(
    const float* __restrict__ x, const float* __restrict__ Watt,
    float* __restrict__ weighted /* [B*T][F] */) {
  const int bt   = blockIdx.x;          // 0..2047
  const int lane = threadIdx.x & 31;
  const int wv   = threadIdx.x >> 5;    // 0..7
  const float* xbt = x + (size_t)bt * (S_ * F_);

  // w_att resident: 16 features per lane (32 lanes * 16 = 512)
  const float* wp = Watt + lane * 16;
  v4f w0 = *(const v4f*)(wp + 0),  w1 = *(const v4f*)(wp + 4);
  v4f w2 = *(const v4f*)(wp + 8),  w3 = *(const v4f*)(wp + 12);

  float m = -INFINITY, l = 0.0f;
  float acc[16];
#pragma unroll
  for (int j = 0; j < 16; ++j) acc[j] = 0.0f;

  for (int s = wv; s < S_; s += 8) {
    const float* xs = xbt + s * F_ + lane * 16;
    if (s + 8 < S_) __builtin_prefetch(xbt + (s + 8) * F_ + lane * 16, 0, 0);
    v4f a0 = __builtin_nontemporal_load((const v4f*)(xs + 0));
    v4f a1 = __builtin_nontemporal_load((const v4f*)(xs + 4));
    v4f a2 = __builtin_nontemporal_load((const v4f*)(xs + 8));
    v4f a3 = __builtin_nontemporal_load((const v4f*)(xs + 12));
    float p = 0.0f;
#pragma unroll
    for (int j = 0; j < 4; ++j)
      p += a0[j] * w0[j] + a1[j] * w1[j] + a2[j] * w2[j] + a3[j] * w3[j];
    p = wave_sum(p);  // att score for slot s, broadcast to all lanes

    float mn = fmaxf(m, p);
    float sc = __expf(m - mn);   // first iter: exp(-inf)=0
    float e  = __expf(p - mn);
    l = l * sc + e;
    m = mn;
#pragma unroll
    for (int j = 0; j < 4; ++j) {
      acc[j]      = acc[j]      * sc + e * a0[j];
      acc[4 + j]  = acc[4 + j]  * sc + e * a1[j];
      acc[8 + j]  = acc[8 + j]  * sc + e * a2[j];
      acc[12 + j] = acc[12 + j] * sc + e * a3[j];
    }
  }

  // merge the 8 per-wave online-softmax states
  __shared__ float sm[8], sl[8];
  __shared__ float sacc[8][F_];
  if (lane == 0) { sm[wv] = m; sl[wv] = l; }
#pragma unroll
  for (int j = 0; j < 16; ++j) sacc[wv][lane * 16 + j] = acc[j];
  __syncthreads();

  float M = -INFINITY;
#pragma unroll
  for (int w = 0; w < 8; ++w) M = fmaxf(M, sm[w]);
  float L = 0.0f, coef[8];
#pragma unroll
  for (int w = 0; w < 8; ++w) { coef[w] = __expf(sm[w] - M); L += coef[w] * sl[w]; }
  const float invL = 1.0f / L;

  for (int f = (int)threadIdx.x; f < F_; f += 256) {
    float v = 0.0f;
#pragma unroll
    for (int w = 0; w < 8; ++w) v += coef[w] * sacc[w][f];
    weighted[(size_t)bt * F_ + f] = v * invL;
  }
}

// ---------------------------------------------------------------------------
// Kernel 2: x_proj = weighted @ W_ih^T + (b_ih + b_hh), via
// v_wmma_f32_16x16x32_f16 (f16 operands, fp32 accumulate).
// One 16x16 output tile per wave; K = 512 -> 16 WMMAs per tile.
// Output stored as [t][gate][b] (b padded stride 8) so the scan kernel's
// C-operand loads are coalesced.
// ---------------------------------------------------------------------------
__global__ __launch_bounds__(256) void xproj_kernel(
    const float* __restrict__ weighted, const float* __restrict__ Wih,
    const float* __restrict__ bih, const float* __restrict__ bhh,
    float* __restrict__ xproj /* [T][G][B] */) {
  const int lane = threadIdx.x & 31;
  const int wgl  = blockIdx.x * 8 + (threadIdx.x >> 5);  // global wave id
  const int Mt   = wgl >> 5;    // 0..127 (row tiles of 2048)
  const int Nt   = wgl & 31;    // 0..31  (col tiles of 512)
  const int r16  = lane & 15;
  const int kg   = lane >> 4;   // K-group select (ISA A/B layout)
  const int col  = Nt * 16 + r16;

  const float* arow = weighted + (size_t)(Mt * 16 + r16) * F_;
  const float* brow = Wih + (size_t)col * F_;  // B[k][n] = W_ih[n][k]

  v8f c = {0.f, 0.f, 0.f, 0.f, 0.f, 0.f, 0.f, 0.f};
#pragma unroll 2
  for (int kb = 0; kb < F_; kb += 32) {
    const int k0 = kb + kg * 8;
    const int k1 = kb + 16 + kg * 8;
    v4f af0 = *(const v4f*)(arow + k0),     af1 = *(const v4f*)(arow + k0 + 4);
    v4f af2 = *(const v4f*)(arow + k1),     af3 = *(const v4f*)(arow + k1 + 4);
    v4f bf0 = *(const v4f*)(brow + k0),     bf1 = *(const v4f*)(brow + k0 + 4);
    v4f bf2 = *(const v4f*)(brow + k1),     bf3 = *(const v4f*)(brow + k1 + 4);
    v16h A, Bv;
#pragma unroll
    for (int j = 0; j < 4; ++j) {
      A[j]       = (_Float16)af0[j];  A[4 + j]  = (_Float16)af1[j];
      A[8 + j]   = (_Float16)af2[j];  A[12 + j] = (_Float16)af3[j];
      Bv[j]      = (_Float16)bf0[j];  Bv[4 + j] = (_Float16)bf1[j];
      Bv[8 + j]  = (_Float16)bf2[j];  Bv[12 + j] = (_Float16)bf3[j];
    }
    c = __builtin_amdgcn_wmma_f32_16x16x32_f16(false, A, false, Bv,
                                               (short)0, c, false, false);
  }

  const float biasv = bih[col] + bhh[col];
#pragma unroll
  for (int r = 0; r < 8; ++r) {
    const int row = Mt * 16 + kg * 8 + r;  // = b*256 + t
    const int b = row >> 8, t = row & 255;
    xproj[((size_t)t * G_ + col) * B_ + b] = c[r] + biasv;
  }
}

// ---------------------------------------------------------------------------
// Kernel 3: LSTM scan (256 sequential steps) + score + masked max.
// Single workgroup of 1024 threads (32 waves). W_hh tile resident in VGPRs
// (f16, 32 VGPRs/lane, loaded once). h kept in LDS as the f16 A-matrix
// (batch 8 padded to M=16 with zeros); c stays fp32 in one register per
// thread (thread <-> (b, j)). 4 WMMAs/wave/step, 2 barriers/step.
// x_proj C-operand is DOUBLE-BUFFERED: loads for step t+1 are issued right
// after step t's WMMA chain, so their latency hides under the barrier +
// cell-update phase instead of sitting on the recurrence critical path.
// sigmoid is monotone -> track max of raw scores, sigmoid once at the end.
// ---------------------------------------------------------------------------
__global__ __launch_bounds__(1024) void lstm_scan_kernel(
    const float* __restrict__ Whh, const float* __restrict__ xproj,
    const float* __restrict__ Wscore, const float* __restrict__ bscore,
    const int* __restrict__ lens, float* __restrict__ out /* [B] */) {
  const int tid  = threadIdx.x;
  const int lane = tid & 31;
  const int wv   = tid >> 5;    // wave id == gate-column tile 0..31
  const int r16  = lane & 15;
  const int kg   = lane >> 4;
  const int bb   = tid >> 7;    // 0..7 : batch for cell update
  const int jj   = tid & 127;   // hidden index for cell update

  __shared__ __align__(16) _Float16 hbuf[16][H_];  // A-matrix (rows 8..15 = 0)
  __shared__ float gates[B_][G_];
  __shared__ float spart[32];
  __shared__ float smax[B_];

  // W_hh tile -> registers (B[k][n] = W_hh[n][k]; row n is contiguous in k)
  const int n = wv * 16 + r16;  // gate row 0..511
  const float* wrow = Whh + (size_t)n * H_;
  v16h Bv[4];
#pragma unroll
  for (int kt = 0; kt < 4; ++kt) {
    const int k0 = kt * 32 + kg * 8, k1 = kt * 32 + 16 + kg * 8;
    v4f f0 = *(const v4f*)(wrow + k0), f1 = *(const v4f*)(wrow + k0 + 4);
    v4f f2 = *(const v4f*)(wrow + k1), f3 = *(const v4f*)(wrow + k1 + 4);
#pragma unroll
    for (int j = 0; j < 4; ++j) {
      Bv[kt][j]      = (_Float16)f0[j];  Bv[kt][4 + j]  = (_Float16)f1[j];
      Bv[kt][8 + j]  = (_Float16)f2[j];  Bv[kt][12 + j] = (_Float16)f3[j];
    }
  }

  float c_state = 0.0f;
  const float ws_j  = Wscore[jj];
  const int   len_t = (tid < B_) ? lens[tid] : 0;
  if (tid < B_) smax[tid] = -INFINITY;
  ((float*)hbuf)[tid] = 0.0f;  // 1024 floats == 2048 halves == entire hbuf

  // prime the x_proj double buffer with t = 0
  const float* xp_base = xproj + (size_t)(wv * 16 + r16) * B_;
  v4f p0 = {0.f, 0.f, 0.f, 0.f}, p1 = {0.f, 0.f, 0.f, 0.f};
  if (kg == 0) {
    p0 = *(const v4f*)(xp_base);
    p1 = *(const v4f*)(xp_base + 4);
  }
  __syncthreads();

  for (int t = 0; t < T_; ++t) {
    // C operand seeded from the preloaded registers (rows 0..7 valid)
    v8f acc = {0.f, 0.f, 0.f, 0.f, 0.f, 0.f, 0.f, 0.f};
    if (kg == 0) {
#pragma unroll
      for (int j = 0; j < 4; ++j) { acc[j] = p0[j]; acc[4 + j] = p1[j]; }
    }

#pragma unroll
    for (int kt = 0; kt < 4; ++kt) {
      const int kb = kt * 32;
      v8h lo = *(const v8h*)&hbuf[r16][kb + kg * 8];
      v8h hi = *(const v8h*)&hbuf[r16][kb + 16 + kg * 8];
      v16h A;
#pragma unroll
      for (int j = 0; j < 8; ++j) { A[j] = lo[j]; A[8 + j] = hi[j]; }
      acc = __builtin_amdgcn_wmma_f32_16x16x32_f16(false, A, false, Bv[kt],
                                                   (short)0, acc, false, false);
    }

    // issue next step's x_proj loads now; latency hides under the barrier
    // and the cell-update phase below
    if (kg == 0 && t + 1 < T_) {
      const float* xp = xp_base + (size_t)(t + 1) * G_ * B_;
      p0 = *(const v4f*)(xp);
      p1 = *(const v4f*)(xp + 4);
    }

    if (kg == 0) {  // lanes 0..15 hold batch rows 0..7 in VGPRs 0..7
      const int col = wv * 16 + r16;
#pragma unroll
      for (int r = 0; r < 8; ++r) gates[r][col] = acc[r];
    }
    __syncthreads();

    // cell update: thread (bb, jj)
    const float iv = sigmoidf_(gates[bb][jj]);
    const float fv = sigmoidf_(gates[bb][128 + jj]);
    const float gv = tanhf(gates[bb][256 + jj]);
    const float ov = sigmoidf_(gates[bb][384 + jj]);
    c_state = fv * c_state + iv * gv;
    const float h = ov * tanhf(c_state);
    hbuf[bb][jj] = (_Float16)h;

    // score partial: wave wv covers batch wv>>2, 32 consecutive j's
    float sp = wave_sum(h * ws_j);
    if (lane == 0) spart[wv] = sp;
    __syncthreads();

    if (tid < B_) {
      const float sc = spart[tid * 4] + spart[tid * 4 + 1] +
                       spart[tid * 4 + 2] + spart[tid * 4 + 3];
      if (t < len_t) smax[tid] = fmaxf(smax[tid], sc);
    }
    // spart/gates for next step are rewritten only after the next barrier;
    // hbuf reads of next step are ordered by the next iteration's barriers.
  }

  __syncthreads();
  if (tid < B_) out[tid] = sigmoidf_(smax[tid] + bscore[0]);
}

// ---------------------------------------------------------------------------
extern "C" void kernel_launch(void* const* d_in, const int* in_sizes, int n_in,
                              void* d_out, int out_size, void* d_ws, size_t ws_size,
                              hipStream_t stream) {
  (void)in_sizes; (void)n_in; (void)out_size; (void)ws_size;
  const float* x      = (const float*)d_in[0];
  const int*   lens   = (const int*)  d_in[1];
  const float* Watt   = (const float*)d_in[2];
  // d_in[3] = b_att: additive constant under softmax -> no-op, skipped
  const float* Wih    = (const float*)d_in[4];
  const float* Whh    = (const float*)d_in[5];
  const float* bih    = (const float*)d_in[6];
  const float* bhh    = (const float*)d_in[7];
  const float* Wscore = (const float*)d_in[8];
  const float* bscore = (const float*)d_in[9];
  float* out = (float*)d_out;

  float* weighted = (float*)d_ws;                    // [2048][512]  (4 MB)
  float* xproj    = weighted + (size_t)B_ * T_ * F_; // [256][512][8] (4 MB)

  attn_pool_kernel<<<B_ * T_, 256, 0, stream>>>(x, Watt, weighted);
  xproj_kernel<<<(128 * 32) / 8, 256, 0, stream>>>(weighted, Wih, bih, bhh, xproj);
  lstm_scan_kernel<<<1, 1024, 0, stream>>>(Whh, xproj, Wscore, bscore, lens, out);
}
